// LocallyDense_39968965656827
// MI455X (gfx1250) — matrix operand order repaired and checked by reading, not compiled, not run on hardware
//
#include <hip/hip_runtime.h>
#include <hip/hip_bf16.h>

// ---------------------------------------------------------------------------
// LocallyDense on MI455X (gfx1250, wave32)
//
// Roofline: ~172 MB of unavoidable HBM traffic (x 64MB + W 86MB + W3 21.5MB)
// vs ~14 GFLOP. bf16 WMMA (f32 accumulate) puts compute under the ~7.4 us
// memory floor, so: stream W and W3 exactly once from HBM, serve the x-gather
// from L2 (x fits in the 192 MB L2), keep WMMA density high per staged byte.
//
// Kernel 1: TILE_M = 256 (all batch rows) so each W column-panel is streamed
//           once. Gather = saddr(x) + 32-bit voffset(rowOff + col*4): one
//           v_add per element, no 64-bit VALU address math.
// Kernel 2: split-K(16) GEMM, global_atomic_add_f32 partial reduction;
//           h-tiles staged by the Tensor Data Mover (pad_enable -> 144B pitch).
// KC=64 per barrier round: 16 WMMAs/wave (k1), 8 WMMAs/wave (k2) per round.
// ---------------------------------------------------------------------------

typedef __bf16 bf16t;
typedef __attribute__((ext_vector_type(16))) __bf16 v16bf;
typedef __attribute__((ext_vector_type(8)))  float  v8f;

typedef __attribute__((ext_vector_type(4))) unsigned int u32x4;
typedef __attribute__((ext_vector_type(8))) int          i32x8;
typedef __attribute__((ext_vector_type(4))) int          i32x4;

#define KC   64
#define LDAP 72   // LDS pitch in bf16: 144B = 16B-aligned, non-pow2 bank stride

#if defined(__has_builtin)
#  if __has_builtin(__builtin_amdgcn_tensor_load_to_lds) && \
      __has_builtin(__builtin_amdgcn_s_wait_tensorcnt)
#    define USE_TDM 1
#  endif
#endif
#ifndef USE_TDM
#  define USE_TDM 0
#endif

#if USE_TDM
#pragma message("CDNA5: TDM path ENABLED (tensor_load_to_lds + s_wait_tensorcnt)")
#else
#pragma message("CDNA5: TDM builtin NOT available; using vector-copy fallback")
#endif

union FragBF {
    v16bf v;
    uint4 q[2];
};

union PackBF {
    bf16t    b[2];
    unsigned u;
};

// CDNA5 16-bit A-matrix 16x32 layout (per 32-deep K-half, selected by kh):
//   lanes 0-15 : row=lane,    elems[0..7]=K0..7,  elems[8..15]=K16..23
//   lanes 16-31: row=lane-16, elems[0..7]=K8..15, elems[8..15]=K24..31
__device__ __forceinline__ v16bf load_a_frag(const bf16t* sA, int rowBase,
                                             int laneRow, int laneHi, int kh) {
    FragBF f;
    const bf16t* p = sA + (rowBase + laneRow) * LDAP + kh * 32 + laneHi * 8;
    f.q[0] = *(const uint4*)(p);
    f.q[1] = *(const uint4*)(p + 16);
    return f.v;
}

// CDNA5 16-bit B-matrix 32x16 layout (B stored transposed in LDS, [col][k]):
//   lanes 0-15 : col=lane,    elems[0..7]=K0..7,  elems[8..15]=K16..23
//   lanes 16-31: col=lane-16, elems[0..7]=K8..15, elems[8..15]=K24..31
__device__ __forceinline__ v16bf load_b_frag(const bf16t* sBT, int colBase,
                                             int laneRow, int laneHi, int kh) {
    FragBF f;
    const bf16t* p = sBT + (colBase + laneRow) * LDAP + kh * 32 + laneHi * 8;
    f.q[0] = *(const uint4*)(p);
    f.q[1] = *(const uint4*)(p + 16);
    return f.v;
}

// ---------------------------------------------------------------------------
// Kernel 1: h[b, n*256+o] = sum_g x[b, idx[n,g]] * W[n,g,o] + bias[n,o]
// Block = (n, 64-wide column panel) covering ALL 256 batch rows: the W panel
// is streamed from HBM exactly once. 41*4 = 164 blocks, 8 waves; wave w owns
// rows [32w, 32w+32) x 64 cols -> 2x4 accumulators, 16 WMMAs per round.
// ---------------------------------------------------------------------------
__global__ __launch_bounds__(256)
void gather_gemm_bf16(const float* __restrict__ x,
                      const int*   __restrict__ gidx,
                      const float* __restrict__ W,
                      const float* __restrict__ bias,
                      bf16t*       __restrict__ hout)
{
    const int D = 65536, G = 2048, O = 256, NO = 41 * 256, TM = 256;

    const int n  = blockIdx.x >> 2;        // expert index 0..40
    const int n0 = (blockIdx.x & 3) * 64;  // output-col panel within O

    __shared__ __align__(16) bf16t sA [TM * LDAP];  // gathered x, [row][k]
    __shared__ __align__(16) bf16t sBT[64 * LDAP];  // W tile transposed [o][k]

    const int tid     = threadIdx.x;
    const int lane    = tid & 31;
    const int wave    = tid >> 5;     // 0..7 -> 32-row slice
    const int laneRow = lane & 15;
    const int laneHi  = lane >> 4;

    // Per-lane row byte offset into x, computed ONCE. rowOff + col*4 stays
    // below 2^31, so gathers are saddr(x) + 32-bit voffset: one v_add each.
    const unsigned rowOff = (unsigned)tid * (unsigned)(D * sizeof(float));

    v8f acc[2][4] = {};

    for (int k0 = 0; k0 < G; k0 += KC) {
        const int gbase = n * G + k0;
        __syncthreads();  // previous round done reading LDS

        // --- stage A: gather 256x64 of x -> bf16 LDS (row = tid) --------
        // col indices are uniform -> s_load_b512 batches; two k packed per
        // ds_store_b32.
        {
            bf16t* pA = &sA[tid * LDAP];
            #pragma unroll
            for (int k = 0; k < KC; k += 2) {
                const unsigned c0 = (unsigned)gidx[gbase + k]     * 4u;
                const unsigned c1 = (unsigned)gidx[gbase + k + 1] * 4u;
                float a0 = *(const float*)((const char*)x + (rowOff + c0));
                float a1 = *(const float*)((const char*)x + (rowOff + c1));
                PackBF p;
                p.b[0] = (bf16t)a0;   // fuses into v_cvt_pk_bf16_f32
                p.b[1] = (bf16t)a1;
                *(unsigned*)(pA + k) = p.u;
            }
        }

        // --- stage B: W[n, k0..+64, n0..+64] -> bf16, transposed --------
        {
            const int o  = tid & 63;          // coalesced over o
            const int kk = (tid >> 6) * 2;    // 0,2,4,6
            const float* Wp = W + ((size_t)n * G + k0) * O + (n0 + o);
            #pragma unroll
            for (int k2 = kk; k2 < KC; k2 += 8) {
                float w0 = Wp[(size_t)k2 * O];
                float w1 = Wp[(size_t)(k2 + 1) * O];
                PackBF p;
                p.b[0] = (bf16t)w0;
                p.b[1] = (bf16t)w1;
                *(unsigned*)&sBT[o * LDAP + k2] = p.u;
            }
            // Speculative prefetch of the next K-chunk (OOB is dropped).
            __builtin_prefetch(Wp + (size_t)(KC + kk) * O, 0, 2);
        }

        __syncthreads();  // tiles visible

        // --- 16 WMMAs per wave per round --------------------------------
        #pragma unroll
        for (int kh = 0; kh < 2; ++kh) {
            v16bf fa0 = load_a_frag(sA, wave * 32,      laneRow, laneHi, kh);
            v16bf fa1 = load_a_frag(sA, wave * 32 + 16, laneRow, laneHi, kh);
            #pragma unroll
            for (int ni = 0; ni < 4; ++ni) {
                v16bf fb = load_b_frag(sBT, ni * 16, laneRow, laneHi, kh);
                acc[0][ni] = __builtin_amdgcn_wmma_f32_16x16x32_bf16(
                                 false, fa0, false, fb, (short)0,
                                 acc[0][ni], false, false);
                acc[1][ni] = __builtin_amdgcn_wmma_f32_16x16x32_bf16(
                                 false, fa1, false, fb, (short)0,
                                 acc[1][ni], false, false);
            }
        }
    }

    // --- epilogue: +bias, store h as bf16 (halves kernel-2 read bytes) ----
    // C/D layout: lanes 0-15 -> N=lane, M=r; lanes 16-31 -> N=lane-16, M=r+8.
    #pragma unroll
    for (int mi = 0; mi < 2; ++mi) {
        const int mbase = wave * 32 + mi * 16 + laneHi * 8;
        #pragma unroll
        for (int ni = 0; ni < 4; ++ni) {
            const int oc = n0 + ni * 16 + laneRow;
            const float bv = bias[n * O + oc];
            const size_t colNO = (size_t)n * O + oc;
            #pragma unroll
            for (int r = 0; r < 8; ++r) {
                float v = acc[mi][ni][r] + bv;
                hout[(size_t)(mbase + r) * NO + colNO] = (bf16t)v;
            }
        }
    }
}

// ---------------------------------------------------------------------------
// Kernel 2: accum += h @ W3  (split-K partial GEMM)
// h bf16 (256 x 10496), W3 f32 (10496 x 512). Tiles 128x64, 16-way split-K:
// 2*8*16 = 256 blocks. Partials reduced with global_atomic_add_f32.
// A-tile staged by the Tensor Data Mover when available (pad_enable gives the
// 144-byte LDS pitch for free), else by vectorized copies.
// ---------------------------------------------------------------------------
__global__ __launch_bounds__(256)
void gemm2_splitk(const bf16t* __restrict__ h,
                  const float* __restrict__ W3,
                  float*       __restrict__ accum)
{
    const int Kdim = 41 * 256, E = 512, TM = 128;
    const int ITERS = Kdim / KC;          // 164
    const int KSPLIT = 16;

    const int bx = blockIdx.x;            // 256 blocks
    const int e0 = (bx & 7) * 64;
    const int m0 = ((bx >> 3) & 1) * TM;
    const int ks = bx >> 4;               // 0..15
    const int it0 = (ITERS * ks) / KSPLIT;
    const int it1 = (ITERS * (ks + 1)) / KSPLIT;

    __shared__ __align__(16) bf16t sA [TM * LDAP];
    __shared__ __align__(16) bf16t sBT[64 * LDAP];

    const int tid     = threadIdx.x;
    const int lane    = tid & 31;
    const int wave    = tid >> 5;     // 0..7 -> 16-row slice
    const int laneRow = lane & 15;
    const int laneHi  = lane >> 4;

    v8f acc[4] = {};

    for (int it = it0; it < it1; ++it) {
        const int k0 = it * KC;
        __syncthreads();

        // --- stage A: h tile (bf16, row-major) ---------------------------
#if USE_TDM
        if (wave == 0) {
            // Tensor DMA descriptor: 2D tile 64 x 128, 2B elems, row stride
            // 10496 elems; LDS padding 16B per 128B row -> 144B pitch (LDAP).
            const unsigned long long ga =
                (unsigned long long)(uintptr_t)(h + (size_t)m0 * Kdim + k0);
            u32x4 g0;
            g0[0] = 1u;                                  // count=1 (valid D#)
            g0[1] = (unsigned)(uintptr_t)&sA[0];         // lds_addr (bytes)
            g0[2] = (unsigned)(ga & 0xFFFFFFFFu);        // global_addr[31:0]
            g0[3] = (unsigned)((ga >> 32) & 0x01FFFFFFu) // global_addr[56:32]
                  | (2u << 30);                          // type = 2 ("image")
            i32x8 g1;
            g1[0] = (1 << 16)        // data_size: 2 bytes
                  | (1 << 20)        // pad_enable
                  | (4 << 22)        // pad_interval: 32 DWORDs (one tile row)
                  | (3 << 25);       // pad_amount: 4 DWORDs (16 bytes)
            g1[1] = (KC << 16);      // tensor_dim0[15:0] = 64
            g1[2] = (TM << 16);      // tensor_dim1[15:0] = 128 (bits 95:80)
            g1[3] = (KC << 16);      // tile_dim0 = 64 (bits 127:112)
            g1[4] = TM;              // tile_dim1 = 128 (bits 143:128)
            g1[5] = Kdim;            // tensor_dim0_stride[31:0] = 10496
            g1[6] = 0;
            g1[7] = 0;
            const i32x4 z4 = {0, 0, 0, 0};
#if defined(__clang_major__) && (__clang_major__ >= 23)
            const i32x8 z8 = {0, 0, 0, 0, 0, 0, 0, 0};
            __builtin_amdgcn_tensor_load_to_lds(g0, g1, z4, z4, z8, 0);
#else
            __builtin_amdgcn_tensor_load_to_lds(g0, g1, z4, z4, 0);
#endif
        }
#else
        {
            const int k4 = (tid & 15) * 4;    // 0,4,...,60
            const int r  = tid >> 4;          // 0..15
            #pragma unroll
            for (int rr = r; rr < TM; rr += 16)
                *(uint2*)&sA[rr * LDAP + k4] =
                    *(const uint2*)&h[(size_t)(m0 + rr) * Kdim + k0 + k4];
        }
#endif

        // --- stage B: W3 tile, f32 -> bf16, transposed (all waves) -------
        {
            const int e  = tid & 63;
            const int kk = (tid >> 6) * 2;    // 0,2,4,6
            const float* W3p = W3 + (size_t)k0 * E + (e0 + e);
            #pragma unroll
            for (int k2 = kk; k2 < KC; k2 += 8) {
                float w0 = W3p[(size_t)k2 * E];
                float w1 = W3p[(size_t)(k2 + 1) * E];
                PackBF p;
                p.b[0] = (bf16t)w0;
                p.b[1] = (bf16t)w1;
                *(unsigned*)&sBT[e * LDAP + k2] = p.u;
            }
            __builtin_prefetch(W3p + (size_t)(KC + kk) * E, 0, 2);
        }

#if USE_TDM
        if (wave == 0) __builtin_amdgcn_s_wait_tensorcnt((short)0);
#endif
        __syncthreads();

        // --- 8 WMMAs per wave per round ----------------------------------
        #pragma unroll
        for (int kh = 0; kh < 2; ++kh) {
            v16bf fa = load_a_frag(sA, wave * 16, laneRow, laneHi, kh);
            #pragma unroll
            for (int ni = 0; ni < 4; ++ni) {
                v16bf fb = load_b_frag(sBT, ni * 16, laneRow, laneHi, kh);
                acc[ni] = __builtin_amdgcn_wmma_f32_16x16x32_bf16(
                              false, fa, false, fb, (short)0, acc[ni],
                              false, false);
            }
        }
    }

    // --- split-K reduction: f32 atomic add into accumulator ----------------
    const int mbase = m0 + wave * 16 + laneHi * 8;
    #pragma unroll
    for (int ni = 0; ni < 4; ++ni) {
        const int ec = e0 + ni * 16 + laneRow;
        #pragma unroll
        for (int r = 0; r < 8; ++r) {
            float* dst = &accum[(size_t)(mbase + r) * E + ec];
#if defined(__HIP_PLATFORM_AMD__)
            unsafeAtomicAdd(dst, acc[ni][r]);   // global_atomic_add_f32
#else
            atomicAdd(dst, acc[ni][r]);
#endif
        }
    }
}

// ---------------------------------------------------------------------------
// Kernel 3: out = leaky_relu(accum + b3), 256x512 f32.
// ---------------------------------------------------------------------------
__global__ __launch_bounds__(256)
void bias_lrelu(const float* __restrict__ accum,
                const float* __restrict__ b3,
                float*       __restrict__ out)
{
    const int i = blockIdx.x * 256 + threadIdx.x;   // 131072 elements
    float v = accum[i] + b3[i & 511];
    out[i] = (v >= 0.0f) ? v : 0.2f * v;
}

// ---------------------------------------------------------------------------
extern "C" void kernel_launch(void* const* d_in, const int* in_sizes, int n_in,
                              void* d_out, int out_size, void* d_ws, size_t ws_size,
                              hipStream_t stream)
{
    (void)in_sizes; (void)n_in; (void)out_size; (void)ws_size;

    const float* x    = (const float*)d_in[0];   // (256, 65536)
    const int*   gidx = (const int*)  d_in[1];   // (41, 2048)
    const float* W    = (const float*)d_in[2];   // (41, 2048, 256)
    const float* b    = (const float*)d_in[3];   // (41, 256)
    const float* W3   = (const float*)d_in[4];   // (10496, 512)
    const float* b3   = (const float*)d_in[5];   // (512,)
    float*       out  = (float*)d_out;           // (256, 512)

    // Workspace layout: [0, 5.125MB) h as bf16; then 512KB f32 accumulator.
    bf16t* hbuf  = (bf16t*)d_ws;                        // 256*10496*2 bytes
    float* accum = (float*)((char*)d_ws + (size_t)256 * 10496 * 2);

    hipMemsetAsync(accum, 0, (size_t)256 * 512 * sizeof(float), stream);

    dim3 blk(256);
    gather_gemm_bf16<<<41 * 4, blk, 0, stream>>>(x, gidx, W, b, hbuf);
    gemm2_splitk    <<<256,    blk, 0, stream>>>(hbuf, W3, accum);
    bias_lrelu      <<<512,    blk, 0, stream>>>(accum, b3, out);
}
